// _FFTConv_14035953123748
// MI455X (gfx1250) — matrix operand order repaired and checked by reading, not compiled, not run on hardware
//
#include <hip/hip_runtime.h>
#include <hip/hip_bf16.h>
#include <stdint.h>

#define B_    16
#define CIN_  64
#define COUT_ 64
#define L_    16384
#define K_    1024
#define PAD_  512
#define TN_   512            // t-positions per workgroup
#define SLEN  1536           // staged signal elements per cin
#define KQ    256            // k elements per quarter slab (32 KB)
#define NBLK  (B_ * (L_ / TN_))   // 512 workgroups

typedef __attribute__((ext_vector_type(16))) __bf16 v16bf;
typedef __attribute__((ext_vector_type(8)))  float  v8f;
typedef __attribute__((ext_vector_type(4)))  int    v4i;

union Frag { uint4 q[2]; unsigned int u[8]; v16bf v; };

__device__ __forceinline__ unsigned short f2bf(float f) {
  unsigned int u = __float_as_uint(f);
  u += 0x7FFFu + ((u >> 16) & 1u);     // round-to-nearest-even
  return (unsigned short)(u >> 16);
}

#if defined(__has_builtin)
#  if __has_builtin(__builtin_amdgcn_global_load_async_to_lds_b128)
#    define HAVE_ASYNC_B128 1
#  endif
#endif

#if HAVE_ASYNC_B128
#define ASYNC_COPY16(g, l)                                         \
  __builtin_amdgcn_global_load_async_to_lds_b128(                  \
      (__attribute__((address_space(1))) v4i*)(g),                 \
      (__attribute__((address_space(3))) v4i*)(l), 0, 0)
#define WAIT_ASYNC() asm volatile("s_wait_asynccnt 0x0" ::: "memory")
#else
#define ASYNC_COPY16(g, l) (*(uint4*)(l) = *(const uint4*)(g))
#define WAIT_ASYNC()
#endif

__global__ void convert_weight(const float* __restrict__ w,
                               unsigned short* __restrict__ wbf, int n) {
  int i = (blockIdx.x * 256 + threadIdx.x) * 4;
  if (i + 3 < n) {
    float4 f = *(const float4*)(w + i);
    ushort4 o;
    o.x = f2bf(f.x); o.y = f2bf(f.y); o.z = f2bf(f.z); o.w = f2bf(f.w);
    *(ushort4*)(wbf + i) = o;
  }
}

__global__ __launch_bounds__(256, 1)
void fftconv_wmma(const float* __restrict__ sig,
                  const unsigned short* __restrict__ wbf,
                  const float* __restrict__ bias,
                  float* __restrict__ out) {
  __shared__ __align__(16) unsigned short wq[2][COUT_ * KQ];   // 2 x 32 KB, double-buffered
  __shared__ __align__(16) unsigned short sig0[SLEN + 16];     // bf16 signal tile
  __shared__ __align__(16) unsigned short sig1[SLEN + 16];     // shifted-by-1 copy

  const int tid  = threadIdx.x;
  const int lane = tid & 31;
  const int wv   = tid >> 5;     // wave 0..7 -> t-range of 64
  const int half = lane >> 4;    // 0 / 1
  const int lm   = lane & 15;

  const int blk = blockIdx.x;
  const int bb  = blk >> 5;              // batch
  const int t0  = (blk & 31) * TN_;      // t-tile origin

  // ---- loop-invariant fragment bases ----
  // A (contiguous k order, same as B): lanes 0-15: K=+0..15; lanes 16-31: K=+16..31
  int arow[4];
#pragma unroll
  for (int ct = 0; ct < 4; ++ct) arow[ct] = ((ct * 16 + lm) << 8) + half * 16;
  // B (contiguous k order, same as A): lane holds column t=lm, 16 consecutive K
  // start parity == lm parity (all k offsets even) -> select copy once
  const unsigned short* bbase[4];
#pragma unroll
  for (int tt = 0; tt < 4; ++tt) {
    int e = wv * 64 + tt * 16 + lm + (half << 4);
    bbase[tt] = (e & 1) ? (sig1 + (e - 1)) : (sig0 + e);
  }

  v8f acc[4][4];
  {
    v8f z = {0.f, 0.f, 0.f, 0.f, 0.f, 0.f, 0.f, 0.f};
#pragma unroll
    for (int i = 0; i < 4; ++i)
#pragma unroll
      for (int j = 0; j < 4; ++j) acc[i][j] = z;
  }

  const float* sb = sig + (size_t)bb * CIN_ * L_;

#pragma unroll 1
  for (int cin = 0; cin < CIN_; ++cin) {
    __syncthreads();   // previous compute done reading sig0/sig1/wq[1]
    // ---- stage bf16 signal row (and +1 shifted copy) for this cin ----
    const float* sc = sb + (size_t)cin * L_;
#pragma unroll 1
    for (int i = tid; i < SLEN; i += 256) {
      int s0 = t0 - PAD_ + i;
      int s1 = s0 + 1;
      float v0 = (s0 >= 0 && s0 < L_) ? sc[s0] : 0.f;
      float v1 = (s1 >= 0 && s1 < L_) ? sc[s1] : 0.f;
      sig0[i] = f2bf(v0);
      sig1[i] = f2bf(v1);
    }

    const unsigned short* wsrc = wbf + (size_t)cin * K_;
    // ---- issue quarter 0 async into wq[0] ----
#pragma unroll
    for (int i = 0; i < 8; ++i) {
      int c  = tid + i * 256;
      int co = c >> 5;
      int kk = (c & 31) << 3;
      ASYNC_COPY16(wsrc + (size_t)co * (CIN_ * K_) + kk, &wq[0][(co << 8) + kk]);
    }

#pragma unroll 1
    for (int q = 0; q < 4; ++q) {
      WAIT_ASYNC();        // quarter q landed in LDS
      __syncthreads();     // visible to all waves; buffer (q+1)&1 free for reuse
      if (q < 3) {
        // issue quarter q+1 while computing quarter q
        const unsigned short* ws2 = wsrc + (q + 1) * KQ;
        unsigned short* dst = wq[(q + 1) & 1];
#pragma unroll
        for (int i = 0; i < 8; ++i) {
          int c  = tid + i * 256;
          int co = c >> 5;
          int kk = (c & 31) << 3;
          ASYNC_COPY16(ws2 + (size_t)co * (CIN_ * K_) + kk, &dst[(co << 8) + kk]);
        }
      }

      const unsigned short* wcur = wq[q & 1];
      const int koff = q * KQ;
#pragma unroll 2
      for (int kc = 0; kc < KQ; kc += 32) {
        Frag afr[4];
#pragma unroll
        for (int ct = 0; ct < 4; ++ct) {
          const unsigned short* p = wcur + arow[ct] + kc;
          afr[ct].q[0] = *(const uint4*)(p);
          afr[ct].q[1] = *(const uint4*)(p + 8);
        }
        Frag bfr[4];
#pragma unroll
        for (int tt = 0; tt < 4; ++tt) {
          const unsigned int* pu = (const unsigned int*)(bbase[tt] + koff + kc);
#pragma unroll
          for (int i = 0; i < 8; ++i) bfr[tt].u[i] = pu[i];
        }
#pragma unroll
        for (int ct = 0; ct < 4; ++ct)
#pragma unroll
          for (int tt = 0; tt < 4; ++tt)
            acc[ct][tt] = __builtin_amdgcn_wmma_f32_16x16x32_bf16(
                false, afr[ct].v, false, bfr[tt].v, (short)0, acc[ct][tt],
                false, false);
      }
    }
  }

  // ---- epilogue: D layout lane=N(t), VGPR j -> M=j (lanes 0-15) / j+8 ----
#pragma unroll
  for (int ct = 0; ct < 4; ++ct) {
#pragma unroll
    for (int j = 0; j < 8; ++j) {
      int cout = ct * 16 + j + half * 8;
      float bv = bias[cout];
      float* ob = out + ((size_t)bb * COUT_ + cout) * L_;
#pragma unroll
      for (int tt = 0; tt < 4; ++tt) {
        int t = t0 + wv * 64 + tt * 16 + lm;
        ob[t] = acc[ct][tt][j] + bv;
      }
    }
  }
}

extern "C" void kernel_launch(void* const* d_in, const int* in_sizes, int n_in,
                              void* d_out, int out_size, void* d_ws, size_t ws_size,
                              hipStream_t stream) {
  (void)in_sizes; (void)n_in; (void)out_size; (void)ws_size;
  const float* sig  = (const float*)d_in[0];
  const float* w    = (const float*)d_in[1];
  const float* bias = (const float*)d_in[2];
  float* out        = (float*)d_out;
  unsigned short* wbf = (unsigned short*)d_ws;   // 8 MB bf16 weight

  const int NW = COUT_ * CIN_ * K_;              // 4,194,304
  convert_weight<<<NW / 1024, 256, 0, stream>>>(w, wbf, NW);
  fftconv_wmma<<<NBLK, 256, 0, stream>>>(sig, wbf, bias, out);
}
